// HackableAttention_7705171329762
// MI455X (gfx1250) — matrix operand order repaired and checked by reading, not compiled, not run on hardware
//
#include <hip/hip_runtime.h>
#include <hip/hip_bf16.h>
#include <stdint.h>

// Problem constants (match reference)
#define B_    2
#define S_    2048
#define D_    2048
#define KVH_  8
#define QPG_  4
#define HD_   64
#define NH_   (KVH_*QPG_)      // 32 query heads
#define NQ_   (NH_*HD_)        // 2048
#define NKV_  (KVH_*HD_)       // 512
#define NTOT_ (NQ_ + 2*NKV_)   // 3072
#define M_    (B_*S_)          // 4096

typedef __attribute__((ext_vector_type(16))) __bf16 v16bf;
typedef __attribute__((ext_vector_type(8)))  float  v8f;

union Frag { uint4 u[2]; v16bf h; };

#ifndef __has_builtin
#define __has_builtin(x) 0
#endif

// CDNA5 async global->LDS path (ASYNCcnt-tracked, bypasses VGPRs).
#if __has_builtin(__builtin_amdgcn_global_load_async_to_lds_b128)
#define ASYNC_LDS 1
#else
#define ASYNC_LDS 0
#endif

#if ASYNC_LDS
// Builtin expects pointers to 16-byte int vectors (per hipcc diagnostic):
//   param0: global (AS1) v4i*, param1: LDS (AS3) v4i*, then imm offset, imm cpol.
typedef int v4i_ __attribute__((vector_size(16)));
typedef __attribute__((address_space(1))) v4i_ v4i_g;
typedef __attribute__((address_space(3))) v4i_ v4i_l;

static __device__ inline void async16(const unsigned short* g, uint4* l) {
  __builtin_amdgcn_global_load_async_to_lds_b128(
      (v4i_g*)(uintptr_t)g,   // global source (strip const via uintptr_t)
      (v4i_l*)l,              // LDS destination (addrspacecast)
      0, 0);
}
#if __has_builtin(__builtin_amdgcn_s_wait_asynccnt)
#define WAIT_ASYNC(n) __builtin_amdgcn_s_wait_asynccnt(n)
#else
#define WAIT_ASYNC(n) asm volatile("s_wait_asynccnt %0" :: "i"(n) : "memory")
#endif
#endif

static __device__ inline unsigned short f2bf(float f) {
  union { float f; unsigned u; } x; x.f = f;
  unsigned r = x.u + 0x7FFFu + ((x.u >> 16) & 1u);   // round-to-nearest-even
  return (unsigned short)(r >> 16);
}

static __device__ inline float redmax16(float x) {
  x = fmaxf(x, __shfl_xor(x, 1, 32));
  x = fmaxf(x, __shfl_xor(x, 2, 32));
  x = fmaxf(x, __shfl_xor(x, 4, 32));
  x = fmaxf(x, __shfl_xor(x, 8, 32));
  return x;
}
static __device__ inline float redsum16(float x) {
  x += __shfl_xor(x, 1, 32);
  x += __shfl_xor(x, 2, 32);
  x += __shfl_xor(x, 4, 32);
  x += __shfl_xor(x, 8, 32);
  return x;
}

// ---------------------------------------------------------------- f32 -> bf16
__global__ __launch_bounds__(256) void cvt_kernel(const float* __restrict__ src,
                                                  unsigned short* __restrict__ dst,
                                                  int n) {
  int i = blockIdx.x * 256 + threadIdx.x;
  if (i < n) dst[i] = f2bf(src[i]);
}

// -------------------------------------------------- 64x64 WMMA tile mainloop
// A: [M][K] bf16 row-major, W: [N][K] bf16 row-major (weights). C += A * W^T.
// 128 threads = 4 waves; wave w computes rows [w*16, w*16+16) of the 64x64 tile.
// Double-buffered LDS; async global->LDS staging when the toolchain exposes it.
static __device__ inline void gemm_tile(const unsigned short* __restrict__ A,
                                        const unsigned short* __restrict__ W,
                                        int m0, int n0, int K, v8f c[4],
                                        uint4 sx[2][256], uint4 sw[2][256]) {
  const int t    = threadIdx.x;
  const int lane = t & 31;
  const int wv   = t >> 5;
  const int lo   = lane & 15;
  const int hi   = lane >> 4;

  #pragma unroll
  for (int f = 0; f < 4; ++f) c[f] = (v8f){0,0,0,0,0,0,0,0};

#if ASYNC_LDS
  // Issue tile k0 into LDS buffer `buf` (4 async b128 per wave: 2 x A, 2 x W).
  #define STAGE_TILE(k0, buf)                                                 \
    do {                                                                      \
      _Pragma("unroll")                                                       \
      for (int i_ = 0; i_ < 2; ++i_) {                                        \
        int linear = t + i_ * 128;                                            \
        int row    = linear >> 2;                                             \
        int chunk  = linear & 3;                                              \
        async16(A + (size_t)(m0 + row) * K + (k0) + chunk * 8,                \
                &sx[buf][linear]);                                            \
        async16(W + (size_t)(n0 + row) * K + (k0) + chunk * 8,                \
                &sw[buf][linear]);                                            \
      }                                                                       \
    } while (0)

  STAGE_TILE(0, 0);                       // prologue
  const int steps = K >> 5;
  for (int it = 0; it < steps; ++it) {
    const int buf = it & 1;
    __syncthreads();                      // all waves done reading buf^1
    if (it + 1 < steps) {
      STAGE_TILE((it + 1) * 32, buf ^ 1); // prefetch next tile into buf^1
      WAIT_ASYNC(4);                      // current tile's 4 asyncs retired
    } else {
      WAIT_ASYNC(0);
    }
    __syncthreads();                      // every wave drained -> tile visible

    Frag a;
    const int ar = wv * 16 + lo;
    a.u[0] = sx[buf][ar * 4 + hi];
    a.u[1] = sx[buf][ar * 4 + 2 + hi];
    #pragma unroll
    for (int f = 0; f < 4; ++f) {
      Frag b;
      const int br = f * 16 + lo;
      b.u[0] = sw[buf][br * 4 + hi];
      b.u[1] = sw[buf][br * 4 + 2 + hi];
      c[f] = __builtin_amdgcn_wmma_f32_16x16x32_bf16(false, a.h, false, b.h,
                                                     (short)0, c[f], false, false);
    }
  }
  #undef STAGE_TILE
#else
  // Fallback: synchronous staging through VGPRs.
  for (int k0 = 0; k0 < K; k0 += 32) {
    __syncthreads();
    #pragma unroll
    for (int i = 0; i < 2; ++i) {
      int linear = t + i * 128;
      int row    = linear >> 2;
      int chunk  = linear & 3;
      const unsigned short* ap = A + (size_t)(m0 + row) * K + k0 + chunk * 8;
      const unsigned short* wp = W + (size_t)(n0 + row) * K + k0 + chunk * 8;
      sx[0][linear] = *(const uint4*)ap;
      sw[0][linear] = *(const uint4*)wp;
      if (k0 + 32 < K) {
        __builtin_prefetch(ap + 32, 0, 1);
        __builtin_prefetch(wp + 32, 0, 1);
      }
    }
    __syncthreads();
    Frag a;
    const int ar = wv * 16 + lo;
    a.u[0] = sx[0][ar * 4 + hi];
    a.u[1] = sx[0][ar * 4 + 2 + hi];
    #pragma unroll
    for (int f = 0; f < 4; ++f) {
      Frag b;
      const int br = f * 16 + lo;
      b.u[0] = sw[0][br * 4 + hi];
      b.u[1] = sw[0][br * 4 + 2 + hi];
      c[f] = __builtin_amdgcn_wmma_f32_16x16x32_bf16(false, a.h, false, b.h,
                                                     (short)0, c[f], false, false);
    }
  }
#endif
}

// ------------------------------------------- QKV projection + RMSNorm + RoPE
__global__ __launch_bounds__(128) void qkv_kernel(
    const unsigned short* __restrict__ xb,
    const unsigned short* __restrict__ wqkv,
    const int*            __restrict__ pos_ids,
    const float*          __restrict__ qn_w,
    const float*          __restrict__ kn_w,
    unsigned short*       __restrict__ qout,   // [B][NH][S][HD]
    unsigned short*       __restrict__ kout,   // [B][KVH][S][HD]
    unsigned short*       __restrict__ vout) { // [B][KVH][HD][S]  (transposed)
  __shared__ uint4 sx[2][256];
  __shared__ uint4 sw[2][256];
  const int t = threadIdx.x, lane = t & 31, wv = t >> 5;
  const int lo = lane & 15, hi = lane >> 4;
  const int m0 = blockIdx.x * 64, n0 = blockIdx.y * 64;

  v8f c[4];
  gemm_tile(xb, wqkv, m0, n0, D_, c, sx, sw);

  const bool isQ = (n0 < NQ_);
  const bool isV = (n0 >= NQ_ + NKV_);

  if (isV) {                                    // V: no norm / no rope, store v^T
    const int hv = (n0 - NQ_ - NKV_) >> 6;
    #pragma unroll
    for (int f = 0; f < 4; ++f) {
      const int hd = f * 16 + lo;
      #pragma unroll
      for (int v = 0; v < 8; ++v) {
        int m = m0 + wv * 16 + v + 8 * hi;
        int b = m >> 11, s = m & (S_ - 1);
        vout[((size_t)(b * KVH_ + hv) * HD_ + hd) * S_ + s] = f2bf(c[f][v]);
      }
    }
    return;
  }

  // RMS norm over the 64-wide head (this tile IS one head)
  float rstd[8];
  #pragma unroll
  for (int v = 0; v < 8; ++v) {
    float ss = c[0][v]*c[0][v] + c[1][v]*c[1][v] + c[2][v]*c[2][v] + c[3][v]*c[3][v];
    ss = redsum16(ss);
    rstd[v] = rsqrtf(ss * (1.0f / 64.0f) + 1e-5f);
  }
  float nv[4][8];
  const float* nwp = isQ ? qn_w : kn_w;
  #pragma unroll
  for (int f = 0; f < 4; ++f) {
    const float nw = nwp[f * 16 + lo];
    #pragma unroll
    for (int v = 0; v < 8; ++v) nv[f][v] = c[f][v] * rstd[v] * nw;
  }

  // Partial RoPE: half=32; only freq indices j<16 rotate (PARTIAL=0.5 of HD)
  #pragma unroll
  for (int f = 0; f < 4; ++f) {
    const int hd = f * 16 + lo;
    const int j  = hd & 31;
    const float invf = (j < 16) ? exp2f(-0.625f * (float)j) : 0.0f;  // 1024^-(j/16)
    #pragma unroll
    for (int v = 0; v < 8; ++v) {
      int m = m0 + wv * 16 + v + 8 * hi;
      int b = m >> 11, s = m & (S_ - 1);
      float ang = (float)pos_ids[s] * invf;
      float sn, cs;
      __sincosf(ang, &sn, &cs);
      float rot = (f < 2) ? -nv[f + 2][v] : nv[f - 2][v];
      float val = nv[f][v] * cs + rot * sn;
      if (isQ) {
        int hq = n0 >> 6;
        qout[((size_t)(b * NH_ + hq) * S_ + s) * HD_ + hd] = f2bf(val);
      } else {
        int hk = (n0 - NQ_) >> 6;
        kout[((size_t)(b * KVH_ + hk) * S_ + s) * HD_ + hd] = f2bf(val);
      }
    }
  }
}

// ------------------------------------------------------------ flash attention
__global__ __launch_bounds__(128) void flash_kernel(
    const unsigned short* __restrict__ q,   // [B][NH][S][HD]
    const unsigned short* __restrict__ k,   // [B][KVH][S][HD]
    const unsigned short* __restrict__ vT,  // [B][KVH][HD][S]
    unsigned short*       __restrict__ att) { // [B*S][NQ_]
  __shared__ uint4 psh[4 * 128];            // 4 waves x 16x64 bf16 P tiles (8KB)
  const int t = threadIdx.x, lane = t & 31, wv = t >> 5;
  const int lo = lane & 15, hi = lane >> 4;
  const int mt = blockIdx.x, h = blockIdx.y, b = blockIdx.z;
  const int kvh = h >> 2;
  const int qbase = mt * 64 + wv * 16;      // this wave's 16 query rows

  const unsigned short* qh = q  + ((size_t)(b * NH_  + h  ) * S_) * HD_;
  const unsigned short* kh = k  + ((size_t)(b * KVH_ + kvh) * S_) * HD_;
  const unsigned short* vh = vT + ((size_t)(b * KVH_ + kvh) * HD_) * S_;

  // Q fragments (kept in registers for the whole kernel)
  Frag qa0, qa1;
  {
    const unsigned short* qrow = qh + (size_t)(qbase + lo) * HD_;
    qa0.u[0] = *(const uint4*)(qrow + hi * 8);
    qa0.u[1] = *(const uint4*)(qrow + 16 + hi * 8);
    qa1.u[0] = *(const uint4*)(qrow + 32 + hi * 8);
    qa1.u[1] = *(const uint4*)(qrow + 48 + hi * 8);
  }

  v8f o[4];
  #pragma unroll
  for (int fo = 0; fo < 4; ++fo) o[fo] = (v8f){0,0,0,0,0,0,0,0};
  float mrow[8], lrow[8];
  #pragma unroll
  for (int v = 0; v < 8; ++v) { mrow[v] = -1e30f; lrow[v] = 0.0f; }

  uint4* pb = psh + wv * 128;
  unsigned short* pbh = (unsigned short*)pb;

  const int kend = qbase + 15;              // causal bound for this strip
  for (int kb = 0; kb <= kend; kb += 64) {
    // ---- scores S = Q K^T  (16 x 64, four 16x16 fragments)
    v8f sf[4];
    #pragma unroll
    for (int f = 0; f < 4; ++f) {
      const unsigned short* krow = kh + (size_t)(kb + f * 16 + lo) * HD_;
      Frag kb0, kb1;
      kb0.u[0] = *(const uint4*)(krow + hi * 8);
      kb0.u[1] = *(const uint4*)(krow + 16 + hi * 8);
      kb1.u[0] = *(const uint4*)(krow + 32 + hi * 8);
      kb1.u[1] = *(const uint4*)(krow + 48 + hi * 8);
      v8f z = (v8f){0,0,0,0,0,0,0,0};
      z = __builtin_amdgcn_wmma_f32_16x16x32_bf16(false, qa0.h, false, kb0.h,
                                                  (short)0, z, false, false);
      z = __builtin_amdgcn_wmma_f32_16x16x32_bf16(false, qa1.h, false, kb1.h,
                                                  (short)0, z, false, false);
      sf[f] = z;
    }
    // ---- scale, tanh soft-cap, causal mask
    #pragma unroll
    for (int f = 0; f < 4; ++f) {
      #pragma unroll
      for (int v = 0; v < 8; ++v) {
        float x = sf[f][v] * 0.125f;                 // 1/sqrt(64)
        x = 50.0f * tanhf(x * 0.02f);                // soft cap
        int sq = qbase + v + 8 * hi;
        int sk = kb + f * 16 + lo;
        sf[f][v] = (sq >= sk) ? x : -1e30f;
      }
    }
    // ---- online softmax update
    float alpha[8];
    #pragma unroll
    for (int v = 0; v < 8; ++v) {
      float tm = fmaxf(fmaxf(sf[0][v], sf[1][v]), fmaxf(sf[2][v], sf[3][v]));
      tm = redmax16(tm);
      float mn = fmaxf(mrow[v], tm);
      alpha[v] = __expf(mrow[v] - mn);
      mrow[v] = mn;
    }
    float rs[8];
    #pragma unroll
    for (int v = 0; v < 8; ++v) rs[v] = 0.0f;
    #pragma unroll
    for (int f = 0; f < 4; ++f) {
      #pragma unroll
      for (int v = 0; v < 8; ++v) {
        float p = __expf(sf[f][v] - mrow[v]);
        sf[f][v] = p;
        rs[v] += p;
      }
    }
    #pragma unroll
    for (int v = 0; v < 8; ++v) {
      lrow[v] = lrow[v] * alpha[v] + redsum16(rs[v]);
    }
    #pragma unroll
    for (int fo = 0; fo < 4; ++fo)
      #pragma unroll
      for (int v = 0; v < 8; ++v) o[fo][v] *= alpha[v];

    // ---- P: C-layout -> A-layout via per-wave LDS round trip (bf16)
    #pragma unroll
    for (int f = 0; f < 4; ++f)
      #pragma unroll
      for (int v = 0; v < 8; ++v)
        pbh[(v + 8 * hi) * 64 + f * 16 + lo] = f2bf(sf[f][v]);

    Frag pa0, pa1;
    pa0.u[0] = pb[lo * 8 + hi];
    pa0.u[1] = pb[lo * 8 + 2 + hi];
    pa1.u[0] = pb[lo * 8 + 4 + hi];
    pa1.u[1] = pb[lo * 8 + 6 + hi];

    // ---- O += P V  (V^T rows give contiguous B fragments)
    #pragma unroll
    for (int fo = 0; fo < 4; ++fo) {
      const unsigned short* vrow = vh + (size_t)(fo * 16 + lo) * S_;
      Frag vb0, vb1;
      vb0.u[0] = *(const uint4*)(vrow + kb + hi * 8);
      vb0.u[1] = *(const uint4*)(vrow + kb + 16 + hi * 8);
      vb1.u[0] = *(const uint4*)(vrow + kb + 32 + hi * 8);
      vb1.u[1] = *(const uint4*)(vrow + kb + 48 + hi * 8);
      o[fo] = __builtin_amdgcn_wmma_f32_16x16x32_bf16(false, pa0.h, false, vb0.h,
                                                      (short)0, o[fo], false, false);
      o[fo] = __builtin_amdgcn_wmma_f32_16x16x32_bf16(false, pa1.h, false, vb1.h,
                                                      (short)0, o[fo], false, false);
    }
  }

  // ---- finalize and store to packed attention buffer [B*S][NH*HD]
  #pragma unroll
  for (int v = 0; v < 8; ++v) {
    const float inv = 1.0f / lrow[v];
    const int s = qbase + v + 8 * hi;
    #pragma unroll
    for (int fo = 0; fo < 4; ++fo) {
      const int n = h * HD_ + fo * 16 + lo;
      att[((size_t)(b * S_ + s)) * NQ_ + n] = f2bf(o[fo][v] * inv);
    }
  }
}

// ------------------------------------------------------------ out projection
__global__ __launch_bounds__(128) void out_kernel(
    const unsigned short* __restrict__ att,  // [M][NQ_] bf16
    const unsigned short* __restrict__ wob,  // [D][NQ_] bf16
    float* __restrict__ out) {               // [M][D] f32
  __shared__ uint4 sx[2][256];
  __shared__ uint4 sw[2][256];
  const int t = threadIdx.x, lane = t & 31, wv = t >> 5;
  const int lo = lane & 15, hi = lane >> 4;
  const int m0 = blockIdx.x * 64, n0 = blockIdx.y * 64;

  v8f c[4];
  gemm_tile(att, wob, m0, n0, NQ_, c, sx, sw);

  #pragma unroll
  for (int f = 0; f < 4; ++f) {
    #pragma unroll
    for (int v = 0; v < 8; ++v) {
      int m = m0 + wv * 16 + v + 8 * hi;
      out[(size_t)m * D_ + n0 + f * 16 + lo] = c[f][v];
    }
  }
}

// ---------------------------------------------------------------------- host
extern "C" void kernel_launch(void* const* d_in, const int* in_sizes, int n_in,
                              void* d_out, int out_size, void* d_ws, size_t ws_size,
                              hipStream_t stream) {
  (void)in_sizes; (void)n_in; (void)out_size; (void)ws_size;
  const float* x       = (const float*)d_in[0];
  const float* wq      = (const float*)d_in[1];
  const float* wk      = (const float*)d_in[2];
  const float* wvp     = (const float*)d_in[3];
  const float* wo      = (const float*)d_in[4];
  const float* qn_w    = (const float*)d_in[5];
  const float* kn_w    = (const float*)d_in[6];
  const int*   pos_ids = (const int*)d_in[7];
  float* out = (float*)d_out;

  char* ws = (char*)d_ws;
  unsigned short* xb   = (unsigned short*)(ws + 0);          // 16.8 MB
  unsigned short* wqkv = (unsigned short*)(ws + 16777216);   // 12.6 MB
  unsigned short* wob  = (unsigned short*)(ws + 29360128);   //  8.4 MB
  unsigned short* qbuf = (unsigned short*)(ws + 37748736);   // 16.8 MB
  unsigned short* kbuf = (unsigned short*)(ws + 54525952);   //  4.2 MB
  unsigned short* vbuf = (unsigned short*)(ws + 58720256);   //  4.2 MB
  unsigned short* attb = (unsigned short*)(ws + 62914560);   // 16.8 MB

  cvt_kernel<<<(M_ * D_ + 255) / 256, 256, 0, stream>>>(x, xb, M_ * D_);
  cvt_kernel<<<(NQ_ * D_ + 255) / 256, 256, 0, stream>>>(wq, wqkv, NQ_ * D_);
  cvt_kernel<<<(NKV_ * D_ + 255) / 256, 256, 0, stream>>>(wk, wqkv + (size_t)NQ_ * D_, NKV_ * D_);
  cvt_kernel<<<(NKV_ * D_ + 255) / 256, 256, 0, stream>>>(wvp, wqkv + (size_t)(NQ_ + NKV_) * D_, NKV_ * D_);
  cvt_kernel<<<(D_ * NQ_ + 255) / 256, 256, 0, stream>>>(wo, wob, D_ * NQ_);

  qkv_kernel<<<dim3(M_ / 64, NTOT_ / 64), 128, 0, stream>>>(
      xb, wqkv, pos_ids, qn_w, kn_w, qbuf, kbuf, vbuf);

  flash_kernel<<<dim3(S_ / 64, NH_, B_), 128, 0, stream>>>(qbuf, kbuf, vbuf, attb);

  out_kernel<<<dim3(M_ / 64, D_ / 64), 128, 0, stream>>>(attb, wob, out);
}
